// DEQ_12498354831496
// MI455X (gfx1250) — matrix operand order repaired and checked by reading, not compiled, not run on hardware
//
#include <hip/hip_runtime.h>
#include <hip/hip_bf16.h>
#include <math.h>

#define N 4096
#define EPS2 1e-8f      // (1e-4)^2
#define MAX_ITERS 30

typedef __attribute__((ext_vector_type(8)))  __bf16 bf16x8;
typedef __attribute__((ext_vector_type(16))) __bf16 bf16x16;
typedef __attribute__((ext_vector_type(8)))  float  f32x8;

// ---------------- setup kernels ----------------

// Jinv = -I in bf16 (32MB). 8 elements per thread via b128 stores.
__global__ void k_init_jinv(__bf16* __restrict__ J) {
    size_t base = ((size_t)blockIdx.x * blockDim.x + threadIdx.x) * 8;
    size_t row = base >> 12, col = base & (N - 1);
    bf16x8 v;
#pragma unroll
    for (int i = 0; i < 8; ++i)
        v[i] = (col + (size_t)i == row) ? (__bf16)(-1.0f) : (__bf16)(0.0f);
    *(bf16x8*)(J + base) = v;
}

// c = U @ x + b   (fp32 matvec, wave-per-row, x staged in LDS)
__global__ __launch_bounds__(256) void k_matvec_c(const float* __restrict__ U,
                                                  const float* __restrict__ x,
                                                  const float* __restrict__ b,
                                                  float* __restrict__ c) {
    __shared__ float xs[N];
    for (int i = threadIdx.x; i < N; i += blockDim.x) xs[i] = x[i];
    __syncthreads();
    int wave = threadIdx.x >> 5, lane = threadIdx.x & 31;
    int row = blockIdx.x * 8 + wave;
    const float* Up = U + (size_t)row * N;
    float acc = 0.f;
    for (int k = lane; k < N; k += 32) acc += Up[k] * xs[k];
    for (int off = 16; off; off >>= 1) acc += __shfl_xor(acc, off, 32);
    if (lane == 0) c[row] = b[row] + acc;
}

// z0 = 0, g0 = tanh(c) (since z0==0), flag = 0
__global__ void k_init_state(const float* __restrict__ c, float* __restrict__ z,
                             float* __restrict__ gz, int* __restrict__ flag) {
    int i = blockIdx.x * blockDim.x + threadIdx.x;
    z[i] = 0.f;
    gz[i] = tanhf(c[i]);
    if (i == 0) *flag = 0;
}

// ---------------- per-iteration kernels (all gated on *flag) ----------------

__global__ void k_tobf(const float* __restrict__ v, __bf16* __restrict__ vbf,
                       const int* __restrict__ flag) {
    if (*flag) return;
    int i = blockIdx.x * blockDim.x + threadIdx.x;
    vbf[i] = (__bf16)v[i];
}

// t = Jinv @ v via v_wmma_f32_16x16x32_bf16. One wave computes 16 rows.
// A fragment: Jinv rows (16-bit A 16x32 layout -> two contiguous b128 loads/lane).
// B fragment: vector broadcast across all 16 columns (staged in LDS).
// MODE 0: dz = -t, z_new = z - t   (alpha = 1).  MODE 1: o0 = t.
template <int MODE>
__global__ __launch_bounds__(128) void k_matvec_wmma(const __bf16* __restrict__ J,
                                                     const __bf16* __restrict__ vbf,
                                                     const float* __restrict__ z,
                                                     float* __restrict__ o0,
                                                     float* __restrict__ o1,
                                                     const int* __restrict__ flag) {
    if (*flag) return;
    __shared__ __bf16 vs[N];                      // 8KB vector stage
    for (int i = threadIdx.x; i < N; i += blockDim.x) vs[i] = vbf[i];
    __syncthreads();
    const int wave = threadIdx.x >> 5, lane = threadIdx.x & 31;
    const int half = lane >> 4, r = lane & 15;
    const int m0 = (blockIdx.x * 4 + wave) * 16;
    const __bf16* Arow = J + (size_t)(m0 + r) * N + half * 8;
    f32x8 acc = {};
    for (int kb = 0; kb < N; kb += 32) {
        bf16x8 alo = *(const bf16x8*)(Arow + kb);
        bf16x8 ahi = *(const bf16x8*)(Arow + kb + 16);
        bf16x16 a = __builtin_shufflevector(alo, ahi, 0,1,2,3,4,5,6,7,8,9,10,11,12,13,14,15);
        const __bf16* bp = vs + kb + half * 8;    // B: lane-half selects K 0..7,16..23 / 8..15,24..31
        bf16x8 blo = *(const bf16x8*)(bp);
        bf16x8 bhi = *(const bf16x8*)(bp + 16);
        bf16x16 b = __builtin_shufflevector(blo, bhi, 0,1,2,3,4,5,6,7,8,9,10,11,12,13,14,15);
        acc = __builtin_amdgcn_wmma_f32_16x16x32_bf16(false, a, false, b,
                                                      (short)0, acc, false, false);
    }
    if (r == 0) {                                  // lanes 0 (M 0..7) and 16 (M 8..15)
        int rowbase = m0 + half * 8;
#pragma unroll
        for (int i = 0; i < 8; ++i) {
            float t = acc[i];
            int row = rowbase + i;
            if (MODE == 0) { o0[row] = -t; o1[row] = z[row] - t; }
            else           { o0[row] = t; }
        }
    }
}

// g_new = tanh(W @ z_new + c) - z_new ; dg_bf = bf16(g_new - gz)
__global__ __launch_bounds__(256) void k_geval(const float* __restrict__ W,
                                               const float* __restrict__ c,
                                               const float* __restrict__ z_new,
                                               const float* __restrict__ gz,
                                               float* __restrict__ g_new,
                                               __bf16* __restrict__ dg_bf,
                                               const int* __restrict__ flag) {
    if (*flag) return;
    __shared__ float zs[N];
    for (int i = threadIdx.x; i < N; i += blockDim.x) zs[i] = z_new[i];
    __syncthreads();
    int wave = threadIdx.x >> 5, lane = threadIdx.x & 31;
    int row = blockIdx.x * 8 + wave;
    const float* Wp = W + (size_t)row * N;
    float acc = 0.f;
    for (int k = lane; k < N; k += 32) acc += Wp[k] * zs[k];
    for (int off = 16; off; off >>= 1) acc += __shfl_xor(acc, off, 32);
    if (lane == 0) {
        float gn = tanhf(acc + c[row]) - zs[row];
        g_new[row] = gn;
        dg_bf[row] = (__bf16)(gn - gz[row]);
    }
}

// single block: denom = dz . Jinv_dg (fixed-order tree), guard, u = (dz - Jinv_dg)/denom
__global__ __launch_bounds__(256) void k_udot(const float* __restrict__ dz,
                                              const float* __restrict__ jdg,
                                              float* __restrict__ u,
                                              const int* __restrict__ flag) {
    if (*flag) return;
    __shared__ float red[256];
    int tid = threadIdx.x;
    float s = 0.f;
    for (int i = tid; i < N; i += 256) s += dz[i] * jdg[i];
    red[tid] = s;
    __syncthreads();
    for (int off = 128; off > 0; off >>= 1) {
        if (tid < off) red[tid] += red[tid + off];
        __syncthreads();
    }
    float denom = red[0];
    if (fabsf(denom) <= 1e-12f) denom = 1e-12f;
    for (int i = tid; i < N; i += 256) u[i] = (dz[i] - jdg[i]) / denom;
}

// vT = dz @ Jinv : deterministic two-stage (row-chunk partials, then fixed-order reduce)
__global__ __launch_bounds__(256) void k_vt_partial(const __bf16* __restrict__ J,
                                                    const float* __restrict__ dz,
                                                    float* __restrict__ partial,
                                                    const int* __restrict__ flag) {
    if (*flag) return;
    int j = blockIdx.x * 256 + threadIdx.x;
    int c0 = blockIdx.y * 128;
    float s = 0.f;
    for (int i = 0; i < 128; ++i)
        s += dz[c0 + i] * (float)J[(size_t)(c0 + i) * N + j];
    partial[(size_t)blockIdx.y * N + j] = s;
}

__global__ void k_vt_reduce(const float* __restrict__ partial, float* __restrict__ vT,
                            const int* __restrict__ flag) {
    if (*flag) return;
    int j = blockIdx.x * blockDim.x + threadIdx.x;
    float s = 0.f;
    for (int cb = 0; cb < 32; ++cb) s += partial[(size_t)cb * N + j];
    vT[j] = s;
}

// Jinv += outer(u, vT) in bf16, 8 columns/thread via b128 RMW
__global__ __launch_bounds__(256) void k_rank1(__bf16* __restrict__ J,
                                               const float* __restrict__ u,
                                               const float* __restrict__ vT,
                                               const int* __restrict__ flag) {
    if (*flag) return;
    size_t idx = (size_t)blockIdx.x * blockDim.x + threadIdx.x;
    int row = (int)(idx >> 9);            // N/8 = 512 groups per row
    int jb  = (int)((idx & 511) << 3);
    __bf16* p = J + (size_t)row * N + jb;
    bf16x8 jv = *(bf16x8*)p;
    f32x8 vv = *(const f32x8*)(vT + jb);
    float ui = u[row];
#pragma unroll
    for (int e = 0; e < 8; ++e) jv[e] = (__bf16)((float)jv[e] + ui * vv[e]);
    *(bf16x8*)p = jv;
}

__global__ void k_commit(const float* __restrict__ z_new, const float* __restrict__ g_new,
                         float* __restrict__ z, float* __restrict__ gz,
                         const int* __restrict__ flag) {
    if (*flag) return;
    int i = blockIdx.x * blockDim.x + threadIdx.x;
    z[i] = z_new[i];
    gz[i] = g_new[i];
}

// last kernel of each iteration: sets flag from ||gz||^2 (cond of the next iteration)
__global__ __launch_bounds__(256) void k_norm_flag(const float* __restrict__ v,
                                                   int* __restrict__ flag) {
    if (*flag) return;
    __shared__ float red[256];
    int tid = threadIdx.x;
    float s = 0.f;
    for (int i = tid; i < N; i += 256) s += v[i] * v[i];
    red[tid] = s;
    __syncthreads();
    for (int off = 128; off > 0; off >>= 1) {
        if (tid < off) red[tid] += red[tid + off];
        __syncthreads();
    }
    if (tid == 0 && red[0] <= EPS2) *flag = 1;
}

// z_star = f(z_root) = z + g(z)
__global__ void k_final(const float* __restrict__ z, const float* __restrict__ gz,
                        float* __restrict__ out) {
    int i = blockIdx.x * blockDim.x + threadIdx.x;
    out[i] = z[i] + gz[i];
}

// ---------------- launcher ----------------

extern "C" void kernel_launch(void* const* d_in, const int* in_sizes, int n_in,
                              void* d_out, int out_size, void* d_ws, size_t ws_size,
                              hipStream_t stream) {
    const float* W = (const float*)d_in[0];
    const float* U = (const float*)d_in[1];
    const float* b = (const float*)d_in[2];
    const float* x = (const float*)d_in[3];
    float* out = (float*)d_out;

    // workspace layout (~32.7 MiB)
    char* w = (char*)d_ws;
    __bf16* Jbf = (__bf16*)w;       w += (size_t)N * N * sizeof(__bf16);   // 32 MiB
    float* partial = (float*)w;     w += (size_t)32 * N * sizeof(float);
    float* c       = (float*)w;     w += N * sizeof(float);
    float* z       = (float*)w;     w += N * sizeof(float);
    float* z_new   = (float*)w;     w += N * sizeof(float);
    float* gz      = (float*)w;     w += N * sizeof(float);
    float* g_new   = (float*)w;     w += N * sizeof(float);
    float* dz      = (float*)w;     w += N * sizeof(float);
    float* jdg     = (float*)w;     w += N * sizeof(float);
    float* u       = (float*)w;     w += N * sizeof(float);
    float* vT      = (float*)w;     w += N * sizeof(float);
    __bf16* gz_bf  = (__bf16*)w;    w += N * sizeof(__bf16);
    __bf16* dg_bf  = (__bf16*)w;    w += N * sizeof(__bf16);
    int* flag      = (int*)w;       w += 256;
    if ((size_t)(w - (char*)d_ws) > ws_size) return;

    k_init_jinv<<<(N / 8) * (N / 256), 256, 0, stream>>>(Jbf);
    k_matvec_c<<<N / 8, 256, 0, stream>>>(U, x, b, c);
    k_init_state<<<N / 256, 256, 0, stream>>>(c, z, gz, flag);
    k_norm_flag<<<1, 256, 0, stream>>>(gz, flag);   // cond before first body

    for (int it = 0; it < MAX_ITERS; ++it) {
        k_tobf<<<N / 256, 256, 0, stream>>>(gz, gz_bf, flag);
        k_matvec_wmma<0><<<64, 128, 0, stream>>>(Jbf, gz_bf, z, dz, z_new, flag);
        k_geval<<<N / 8, 256, 0, stream>>>(W, c, z_new, gz, g_new, dg_bf, flag);
        k_matvec_wmma<1><<<64, 128, 0, stream>>>(Jbf, dg_bf, nullptr, jdg, nullptr, flag);
        k_udot<<<1, 256, 0, stream>>>(dz, jdg, u, flag);
        k_vt_partial<<<dim3(16, 32), 256, 0, stream>>>(Jbf, dz, partial, flag);
        k_vt_reduce<<<N / 256, 256, 0, stream>>>(partial, vT, flag);
        k_rank1<<<(N * (N / 8)) / 256, 256, 0, stream>>>(Jbf, u, vT, flag);
        k_commit<<<N / 256, 256, 0, stream>>>(z_new, g_new, z, gz, flag);
        k_norm_flag<<<1, 256, 0, stream>>>(gz, flag);  // cond for next iteration
    }

    k_final<<<N / 256, 256, 0, stream>>>(z, gz, out);
}
